// DSQGAttentionQW_66374424592425
// MI455X (gfx1250) — matrix operand order repaired and checked by reading, not compiled, not run on hardware
//
#include <hip/hip_runtime.h>
#include <hip/hip_bf16.h>

// ---------------------------------------------------------------------------
// DSQG attention for MI455X (gfx1250, wave32).
//   Matrices are pre-converted ONCE to bf16 in WMMA fragment order in global
//   memory; GEMMs load fragments straight to VGPRs (global_load_b128 x2 per
//   fragment, no LDS, no barriers) and run v_wmma_f32_16x16x32_bf16.
//   Wave tile = 32x64 (2x4 WMMA tiles) -> 6 fragment loads per 8 WMMAs.
//   47-offset gather-softmax attention stays on wave32 VALU + shfl.
// ---------------------------------------------------------------------------

typedef __attribute__((ext_vector_type(16))) __bf16          v16bf;
typedef __attribute__((ext_vector_type(16))) unsigned short  v16us;
typedef __attribute__((ext_vector_type(8)))  float           v8f;

#define NOFF 47
#define HEADS 8
#define HD 64
#define DMODEL 512
#define SEQ 2048

__constant__ int c_offsets[NOFF] = {
    0, 1, 2, 3, 4, 5, 6, 7, 8, 9, 10, 11, 12, 13, 14, 15, 16,
    17, 18, 19, 20, 21, 22, 23, 24, 25, 26, 27, 28, 29, 30, 31, 32,
    48, 64, 96, 128, 192, 256, 384, 512, 768, 1024, 1536, 2048, 3072, 4096};

__device__ __forceinline__ unsigned short f2bf(float f) {
  unsigned u = __builtin_bit_cast(unsigned, f);
  unsigned r = u + 0x7FFFu + ((u >> 16) & 1u);  // round-to-nearest-even
  return (unsigned short)(r >> 16);
}

__device__ __forceinline__ unsigned long long pack4(float x, float y, float z, float w) {
  return (unsigned long long)f2bf(x)         |
         ((unsigned long long)f2bf(y) << 16) |
         ((unsigned long long)f2bf(z) << 32) |
         ((unsigned long long)f2bf(w) << 48);
}

__device__ __forceinline__ v8f wmma_bf16(v16us au, v16us bu, v8f c) {
  v16bf a = __builtin_bit_cast(v16bf, au);
  v16bf b = __builtin_bit_cast(v16bf, bu);
  return __builtin_amdgcn_wmma_f32_16x16x32_bf16(
      false, a, false, b, (short)0, c, false, false);
}

// ---------------------------------------------------------------------------
// Fragment-order global layouts (ISA 7.12.2, bf16, wave32):
//  A-frag: [mSub = r/16][kb = c/32][lane][e], lane = (r&15)+16*((c>>3)&1),
//          e = (c&7)+8*(c>>4)        (one 16x32 tile = 32 lanes x 16 bf16)
//  B-frag: [nSub = c/16][kb = r/32][lane][e], lane = 16*((r>>3)&1)+(c&15),
//          e = (r&7)+8*(r>>4)                  (one 32x16 tile = 32 lanes x 16 bf16)
// ---------------------------------------------------------------------------

// fp32 [R x C] (optionally .* mul) -> A-fragment bf16. One thread = 4 elems.
template <bool FUSE>
__global__ __launch_bounds__(256) void convA_kernel(
    const float* __restrict__ src, const float* __restrict__ mul,
    unsigned short* __restrict__ dst, int R, int C) {
  const int g    = blockIdx.x * 256 + threadIdx.x;   // R*C/4 threads
  const int e0   = (g & 3) * 4;                      // 0,4,8,12
  const int lane = (g >> 2) & 31;
  const int tile = g >> 7;                           // 128 threads per tile
  const int KB   = C >> 5;
  const int mSub = tile / KB, kb = tile % KB;
  const int row  = mSub * 16 + (lane & 15);
  const int half = lane >> 4;
  // invert e -> c: c = (e&7) + 16*(e>>3) + 8*half ; e0..e0+3 are contiguous c
  const int c0 = (e0 & 7) + 16 * (e0 >> 3) + 8 * half;
  const size_t si = (size_t)row * C + kb * 32 + c0;
  float4 v = *(const float4*)&src[si];
  if (FUSE) {
    const float4 m = *(const float4*)&mul[si];
    v.x *= m.x; v.y *= m.y; v.z *= m.z; v.w *= m.w;
  }
  *(unsigned long long*)&dst[(size_t)tile * 512 + lane * 16 + e0] =
      pack4(v.x, v.y, v.z, v.w);
}

// fp32 [K x N] -> B-fragment bf16. One thread = 4 elems (4 rows, same col).
__global__ __launch_bounds__(256) void convB_kernel(
    const float* __restrict__ src, unsigned short* __restrict__ dst,
    int Kr, int Nc) {
  const int g    = blockIdx.x * 256 + threadIdx.x;   // K*N/4 threads
  const int e0   = (g & 3) * 4;
  const int lane = (g >> 2) & 31;
  const int tile = g >> 7;
  const int KB   = Kr >> 5;
  const int nSub = tile / KB, kb = tile % KB;
  const int col  = nSub * 16 + (lane & 15);
  const int r0   = kb * 32 + (e0 & 7) + 16 * (e0 >> 3) + 8 * (lane >> 4);
  const float x0 = src[(size_t)(r0 + 0) * Nc + col];
  const float x1 = src[(size_t)(r0 + 1) * Nc + col];
  const float x2 = src[(size_t)(r0 + 2) * Nc + col];
  const float x3 = src[(size_t)(r0 + 3) * Nc + col];
  *(unsigned long long*)&dst[(size_t)tile * 512 + lane * 16 + e0] =
      pack4(x0, x1, x2, x3);
}

// ---------------------------------------------------------------------------
// GEMM on fragment-order operands: C[M,N] = act(Afrag @ Bfrag + bias[N]).
// 128 threads = 4 waves (2x2); wave tile 32x64 = 2x4 WMMA tiles.
// Block tile 64x128. No LDS, no barriers: fragments come straight from
// global (WGP$/L2-resident) via 32-byte vector loads.
// ---------------------------------------------------------------------------
template <bool SIG>
__global__ __launch_bounds__(128) void gemm_frag_wmma_kernel(
    const unsigned short* __restrict__ aG,  // [M/16][K/32][32][16]
    const unsigned short* __restrict__ bG,  // [N/16][K/32][32][16]
    const float* __restrict__ bias,
    float* __restrict__ C, int M, int N, int K) {
  const int lane = threadIdx.x & 31;
  const int wave = threadIdx.x >> 5;
  const int wm   = wave >> 1;            // 0..1
  const int wn   = wave & 1;             // 0..1
  const int half = lane >> 4;
  const int l15  = lane & 15;
  const int KB   = K >> 5;

  const int mSub0 = blockIdx.x * 4 + 2 * wm;   // 2 A-subtiles for this wave
  const int nSub0 = blockIdx.y * 8 + 4 * wn;   // 4 B-subtiles for this wave

  const unsigned short* aP0 = aG + ((size_t)(mSub0 + 0) * KB) * 512 + lane * 16;
  const unsigned short* aP1 = aG + ((size_t)(mSub0 + 1) * KB) * 512 + lane * 16;
  const unsigned short* bP0 = bG + ((size_t)(nSub0 + 0) * KB) * 512 + lane * 16;
  const unsigned short* bP1 = bG + ((size_t)(nSub0 + 1) * KB) * 512 + lane * 16;
  const unsigned short* bP2 = bG + ((size_t)(nSub0 + 2) * KB) * 512 + lane * 16;
  const unsigned short* bP3 = bG + ((size_t)(nSub0 + 3) * KB) * 512 + lane * 16;

  v8f acc[2][4] = {};

  for (int kb = 0; kb < KB; ++kb) {
    const size_t o = (size_t)kb * 512;
    // Prefetch two K-blocks ahead (global_prefetch_b8).
    __builtin_prefetch(aP0 + o + 1024, 0, 1);
    __builtin_prefetch(bP0 + o + 1024, 0, 1);

    v16us a0 = *(const v16us*)(aP0 + o);
    v16us a1 = *(const v16us*)(aP1 + o);
    v16us b0 = *(const v16us*)(bP0 + o);
    v16us b1 = *(const v16us*)(bP1 + o);
    v16us b2 = *(const v16us*)(bP2 + o);
    v16us b3 = *(const v16us*)(bP3 + o);

    acc[0][0] = wmma_bf16(a0, b0, acc[0][0]);
    acc[1][0] = wmma_bf16(a1, b0, acc[1][0]);
    acc[0][1] = wmma_bf16(a0, b1, acc[0][1]);
    acc[1][1] = wmma_bf16(a1, b1, acc[1][1]);
    acc[0][2] = wmma_bf16(a0, b2, acc[0][2]);
    acc[1][2] = wmma_bf16(a1, b2, acc[1][2]);
    acc[0][3] = wmma_bf16(a0, b3, acc[0][3]);
    acc[1][3] = wmma_bf16(a1, b3, acc[1][3]);
  }

  // Epilogue: C/D layout -> VGPR r, lane l: M = r + 8*(l>=16), N = l&15.
  #pragma unroll
  for (int j = 0; j < 4; ++j) {
    const int col = (nSub0 + j) * 16 + l15;
    const float bv = bias[col];
    #pragma unroll
    for (int i = 0; i < 2; ++i) {
      const int rowB = (mSub0 + i) * 16 + half * 8;
      #pragma unroll
      for (int r = 0; r < 8; ++r) {
        float v = acc[i][j][r] + bv;
        if (SIG) v = 1.0f / (1.0f + __expf(-v));
        C[(size_t)(rowB + r) * N + col] = v;
      }
    }
  }
}

// ---------------------------------------------------------------------------
// Attention: one wave32 per (position, head). Each lane owns 2 of the 64
// head dims; 47 offset scores via cross-lane reduction, softmax, weighted v.
// ---------------------------------------------------------------------------
__global__ __launch_bounds__(128) void dsqg_attn_kernel(
    const float* __restrict__ qkv,         // [N, 3*D] rows: q | k | v
    const float* __restrict__ pos_bias,    // [NOFF, H]
    const float* __restrict__ scale_embed, // [NOFF, HD]
    const float* __restrict__ if_gain,     // [H]
    const float* __restrict__ disp_amp,    // [H]
    float* __restrict__ out)               // [N, D]
{
  const int wave = threadIdx.x >> 5;
  const int lane = threadIdx.x & 31;
  const int n = blockIdx.x * 4 + wave;
  const int h = blockIdx.y;
  const int d0 = lane * 2;
  const int d1 = d0 + 1;

  const float* qrow = qkv + (size_t)n * (3 * DMODEL) + h * HD;
  const float q0 = qrow[d0];
  const float q1 = qrow[d1];

  // Head dispersion frequency: {0,0,w,w,4w,4w,6w,6w}, w = pi / log(1+4096)
  const float wbase = 3.14159265358979f / __logf(4097.0f);
  const int mult = (h >= 6) ? 6 : (h >= 4) ? 4 : (h >= 2) ? 1 : 0;
  const float omg = wbase * (float)mult;
  const float da  = disp_amp[h];

  float s[NOFF];
  #pragma unroll 1
  for (int o = 0; o < NOFF; ++o) {
    const int off = c_offsets[o];
    const int j = n - off;
    float sc = -3.0e38f;
    if (j >= 0) {
      const float* krow = qkv + (size_t)j * (3 * DMODEL) + DMODEL + h * HD;
      float se0 = 1.0f + scale_embed[o * HD + d0];
      float se1 = 1.0f + scale_embed[o * HD + d1];
      float p = q0 * krow[d0] * se0 + q1 * krow[d1] * se1;
      #pragma unroll
      for (int i = 16; i > 0; i >>= 1) p += __shfl_xor(p, i, 32);
      float ck = (omg > 0.0f) ? __cosf(omg * __logf(1.0f + (float)off)) : 0.0f;
      sc = p * 0.125f /* 1/sqrt(64) */ + pos_bias[o * HEADS + h] + ck * da;
    }
    s[o] = sc;
  }

  float mx = -3.0e38f;
  #pragma unroll 1
  for (int o = 0; o < NOFF; ++o) mx = fmaxf(mx, s[o]);
  float Z = 0.0f;
  #pragma unroll 1
  for (int o = 0; o < NOFF; ++o) { s[o] = __expf(s[o] - mx); Z += s[o]; }
  const float inv = 1.0f / Z;

  float a0 = 0.0f, a1 = 0.0f;
  #pragma unroll 1
  for (int o = 0; o < NOFF; ++o) {
    const int off = c_offsets[o];
    const int j = n - off;
    if (j >= 0) {
      const float* vrow = qkv + (size_t)j * (3 * DMODEL) + 2 * DMODEL + h * HD;
      const float p = s[o] * inv;
      a0 += p * vrow[d0];
      a1 += p * vrow[d1];
    }
  }

  const float g = if_gain[h];
  float* orow = out + (size_t)n * DMODEL + h * HD;
  orow[d0] = a0 * g;
  orow[d1] = a1 * g;
}

// ---------------------------------------------------------------------------
// Launch
// ---------------------------------------------------------------------------
extern "C" void kernel_launch(void* const* d_in, const int* in_sizes, int n_in,
                              void* d_out, int out_size, void* d_ws, size_t ws_size,
                              hipStream_t stream) {
  const float* x           = (const float*)d_in[0];   // [N, D]
  const float* W_qkv       = (const float*)d_in[1];   // [D, 3D]
  const float* b_qkv       = (const float*)d_in[2];   // [3D]
  const float* W_out       = (const float*)d_in[3];   // [D, D]
  const float* b_out       = (const float*)d_in[4];   // [D]
  const float* W_gate      = (const float*)d_in[5];   // [D, D]
  const float* b_gate      = (const float*)d_in[6];   // [D]
  const float* pos_bias    = (const float*)d_in[7];   // [NOFF, H]
  const float* scale_embed = (const float*)d_in[8];   // [NOFF, HD]
  const float* if_gain     = (const float*)d_in[9];   // [H]
  const float* disp_amp    = (const float*)d_in[10];  // [H]
  float* outp = (float*)d_out;

  // ---- Workspace layout ----
  float* qkv  = (float*)d_ws;                        // [N, 3D]  fp32
  float* gate = qkv + (size_t)SEQ * 3 * DMODEL;      // [N, D]   fp32
  float* aout = gate + (size_t)SEQ * DMODEL;         // [N, D]   fp32
  unsigned short* xF     = (unsigned short*)(aout + (size_t)SEQ * DMODEL);
  unsigned short* aoutF  = xF + (size_t)SEQ * DMODEL;          // A-frags
  unsigned short* wqkvF  = aoutF + (size_t)SEQ * DMODEL;       // B-frags
  unsigned short* wgateF = wqkvF + (size_t)DMODEL * 3 * DMODEL;
  unsigned short* woutF  = wgateF + (size_t)DMODEL * DMODEL;

  // ---- One-time fp32 -> bf16 fragment-order conversions ----
  convA_kernel<false><<<SEQ * DMODEL / 4 / 256, 256, 0, stream>>>(
      x, nullptr, xF, SEQ, DMODEL);
  convB_kernel<<<DMODEL * 3 * DMODEL / 4 / 256, 256, 0, stream>>>(
      W_qkv, wqkvF, DMODEL, 3 * DMODEL);
  convB_kernel<<<DMODEL * DMODEL / 4 / 256, 256, 0, stream>>>(
      W_gate, wgateF, DMODEL, DMODEL);
  convB_kernel<<<DMODEL * DMODEL / 4 / 256, 256, 0, stream>>>(
      W_out, woutF, DMODEL, DMODEL);

  dim3 blk(128);

  // 1) qkv = x @ W_qkv + b_qkv           (2048 x 512 x 1536)
  gemm_frag_wmma_kernel<false>
      <<<dim3(SEQ / 64, (3 * DMODEL) / 128), blk, 0, stream>>>(
          xF, wqkvF, b_qkv, qkv, SEQ, 3 * DMODEL, DMODEL);

  // 2) gate = sigmoid(x @ W_gate + b_gate)   (2048 x 512 x 512)
  gemm_frag_wmma_kernel<true>
      <<<dim3(SEQ / 64, DMODEL / 128), blk, 0, stream>>>(
          xF, wgateF, b_gate, gate, SEQ, DMODEL, DMODEL);

  // 3) dilated-offset attention -> aout (fp32)
  dsqg_attn_kernel<<<dim3(SEQ / 4, HEADS), blk, 0, stream>>>(
      qkv, pos_bias, scale_embed, if_gain, disp_amp, aout);

  // 4) aoutF = frag(aout .* gate); out = aoutF @ W_out + b_out
  convA_kernel<true><<<SEQ * DMODEL / 4 / 256, 256, 0, stream>>>(
      aout, gate, aoutF, SEQ, DMODEL);
  gemm_frag_wmma_kernel<false>
      <<<dim3(SEQ / 64, DMODEL / 128), blk, 0, stream>>>(
          aoutF, woutF, b_out, outp, SEQ, DMODEL, DMODEL);
}
